// OptimizedGPT2Attention_71966472012549
// MI455X (gfx1250) — compile-verified, matrix operand order checked by reading, not run on hardware
//
#include <hip/hip_runtime.h>

typedef __attribute__((ext_vector_type(16))) __bf16 v16bf;
typedef __attribute__((ext_vector_type(8)))  __bf16 v8bf;
typedef __attribute__((ext_vector_type(8)))  float  v8f;
typedef __attribute__((ext_vector_type(4)))  unsigned int v4u;
typedef __attribute__((ext_vector_type(8)))  int v8i;
typedef __attribute__((ext_vector_type(4)))  int v4i;

#define D_MODEL 2048
#define NHEAD   16
#define HD      128
#define SEQ     2048
#define BATCH   4
#define ROWS    (BATCH*SEQ)   // 8192

// ---------------------------------------------------------------------------
// WMMA helper: D = A(16x32 bf16) * B(32x16 bf16) + C(16x16 f32)
// ---------------------------------------------------------------------------
__device__ __forceinline__ v8f wmma_bf16(v16bf a, v16bf b, v8f c) {
  return __builtin_amdgcn_wmma_f32_16x16x32_bf16(
      /*neg_a=*/false, a, /*neg_b=*/false, b,
      /*c_mod=*/(short)0, c, /*reuse_a=*/false, /*reuse_b=*/false);
}

// Per-lane fragment row for any 16-bit 16x32 operand (A-style, or B read from
// a transposed tile). p = &tile[row*stride + half*8]; elements 0..7 are
// K = {8*half..}, elements 8..15 are K = {16+8*half..} -> two 16B loads.
__device__ __forceinline__ v16bf ld_frag(const __bf16* p) {
  v8bf lo = *(const v8bf*)(p);
  v8bf hi = *(const v8bf*)(p + 16);
  v16bf r;
#pragma unroll
  for (int i = 0; i < 8; ++i) { r[i] = lo[i]; r[i + 8] = hi[i]; }
  return r;
}

// ---------------------------------------------------------------------------
// Tensor Data Mover: 2D tile (tile_d0 contiguous elems x tile_d1 rows of
// stride0 elems, 2-byte data) global -> LDS, with per-row LDS padding.
// pad_interval/pad_amount encoded per D# group1 (row bytes == interval, so
// padding is a pure address skip after each row -> LDS row stride
// = tile_d0 + 8 bf16). Descriptor built per CDNA5 ISA 8.3/8.4.
// Issue from ONE wave; consumer side: s_wait_tensorcnt 0 + barrier.
// This toolchain exposes the 6-arg builtin (extra int32x8 group before cpol).
// ---------------------------------------------------------------------------
__device__ __forceinline__ void tdm_load_2d(
    unsigned lds_off, const void* gptr,
    unsigned tile_d0, unsigned tile_d1,
    unsigned long long stride0_elems,
    unsigned pad_interval, unsigned pad_amount) {
  unsigned long long ga = (unsigned long long)(uintptr_t)gptr;
  v4u g0;
  g0[0] = 1u;                                              // count=1, user mode
  g0[1] = lds_off;                                         // LDS byte address
  g0[2] = (unsigned)(ga & 0xffffffffull);                  // global_addr[31:0]
  g0[3] = (unsigned)((ga >> 32) & 0x1ffffffull)            // global_addr[56:32]
        | 0x80000000u;                                     // type=2 ("image")
  v8i g1;
  g1[0] = (int)((1u << 16)                                 // data_size = 2B
        | (1u << 20)                                       // pad_enable
        | (pad_interval << 22) | (pad_amount << 25));
  g1[1] = (int)((tile_d0 & 0xffffu) << 16);                // tensor_dim0[15:0]
  g1[2] = (int)((tile_d0 >> 16) | ((tile_d1 & 0xffffu) << 16)); // td0 hi, td1 lo
  g1[3] = (int)((tile_d1 >> 16) | (tile_d0 << 16));        // td1 hi, tile_dim0
  g1[4] = (int)(tile_d1 & 0xffffu);                        // tile_dim1 (dim2=0)
  g1[5] = (int)(stride0_elems & 0xffffffffull);            // dim0_stride[31:0]
  g1[6] = (int)((stride0_elems >> 32) & 0xffffull);        // dim0_stride[47:32]
  g1[7] = 0;
  v4i g2 = {0, 0, 0, 0};                                   // 2D: dims 2/3 unused
  v4i g3 = {0, 0, 0, 0};
  v8i g4 = {0, 0, 0, 0, 0, 0, 0, 0};                       // unused extra group
  __builtin_amdgcn_tensor_load_to_lds(g0, g1, g2, g3, g4, 0);
}

// ---------------------------------------------------------------------------
// Transpose + convert: W[R][C] fp32 -> Wt[C][R] bf16 (32x32 LDS tiles)
// ---------------------------------------------------------------------------
__global__ __launch_bounds__(256) void transpose_cvt_kernel(
    const float* __restrict__ src, __bf16* __restrict__ dst, int R, int C) {
  __shared__ float tile[32][33];
  const int c0 = blockIdx.x * 32, r0 = blockIdx.y * 32;
  const int tx = threadIdx.x & 31, ty = threadIdx.x >> 5;   // 32 x 8
#pragma unroll
  for (int i = 0; i < 32; i += 8)
    tile[tx][ty + i] = src[(size_t)(r0 + ty + i) * C + c0 + tx];
  __syncthreads();
#pragma unroll
  for (int i = 0; i < 32; i += 8)
    dst[(size_t)(c0 + ty + i) * R + r0 + tx] = (__bf16)tile[ty + i][tx];
}

// ---------------------------------------------------------------------------
// LayerNorm: one block per row, emits bf16
// ---------------------------------------------------------------------------
__global__ __launch_bounds__(256) void ln_bf16_kernel(
    const float* __restrict__ x, const float* __restrict__ g,
    const float* __restrict__ bta, __bf16* __restrict__ out) {
  __shared__ float red[16];
  const int row = blockIdx.x;
  const int tid = threadIdx.x;
  const float* xr = x + (size_t)row * D_MODEL;
  float4 v0 = ((const float4*)xr)[tid*2];
  float4 v1 = ((const float4*)xr)[tid*2 + 1];
  float xs[8] = {v0.x, v0.y, v0.z, v0.w, v1.x, v1.y, v1.z, v1.w};
  float s = 0.f, s2 = 0.f;
#pragma unroll
  for (int i = 0; i < 8; ++i) { s += xs[i]; s2 += xs[i]*xs[i]; }
#pragma unroll
  for (int m = 16; m >= 1; m >>= 1) {
    s  += __shfl_xor(s,  m, 32);
    s2 += __shfl_xor(s2, m, 32);
  }
  if ((tid & 31) == 0) { red[tid >> 5] = s; red[8 + (tid >> 5)] = s2; }
  __syncthreads();
  if (tid == 0) {
    float ts = 0.f, ts2 = 0.f;
#pragma unroll
    for (int i = 0; i < 8; ++i) { ts += red[i]; ts2 += red[8 + i]; }
    red[0] = ts; red[8] = ts2;
  }
  __syncthreads();
  const float mean = red[0] * (1.0f / D_MODEL);
  const float var  = red[8] * (1.0f / D_MODEL) - mean*mean;
  const float rstd = rsqrtf(var + 1e-5f);
  const int base = tid * 8;
#pragma unroll
  for (int i = 0; i < 8; ++i) {
    const int j = base + i;
    out[(size_t)row * D_MODEL + j] = (__bf16)((xs[i] - mean) * rstd * g[j] + bta[j]);
  }
}

// ---------------------------------------------------------------------------
// Tiled bf16 WMMA GEMM: C[M,N] = A[M,K] * Bt[N,K]^T + bias
// Block tile 128x128x64, TDM-staged tiles (LDS stride 72), 8 waves = 32x64.
// MODE 0: QKV epilogue -> Q/K bf16 [B,H,S,hd] (Q pre-scaled), V^T [B,H,hd,S]
// MODE 1: fp32 output (final projection)
// ---------------------------------------------------------------------------
template <int MODE>
__global__ __launch_bounds__(256) void gemm_bf16_kernel(
    const __bf16* __restrict__ A, const __bf16* __restrict__ Bt,
    const float* __restrict__ bias, int M, int N, int K,
    __bf16* __restrict__ Qb, __bf16* __restrict__ Kb, __bf16* __restrict__ Vt,
    float* __restrict__ outf) {
  __shared__ __bf16 As[128 * 72];   // [m][k], TDM-padded stride 72 (144B)
  __shared__ __bf16 Bs[128 * 72];   // [n][k], TDM-padded stride 72
  const int tid  = threadIdx.x;
  const int lane = tid & 31, wv = tid >> 5;
  const int lidx = lane & 15, hf = lane >> 4;
  const int wm = wv >> 1, wn = wv & 1;
  const int m0 = blockIdx.y * 128;
  const int n0 = blockIdx.x * 128;

  v8f acc[2][4] = {};

  for (int k0 = 0; k0 < K; k0 += 64) {
    if (wv == 0) {  // one wave drives the Tensor Data Mover
      tdm_load_2d((unsigned)(uintptr_t)&As[0], A  + (size_t)m0*K + k0,
                  64, 128, (unsigned long long)K, 4, 3);   // 128B rows -> +16B pad
      tdm_load_2d((unsigned)(uintptr_t)&Bs[0], Bt + (size_t)n0*K + k0,
                  64, 128, (unsigned long long)K, 4, 3);
      __builtin_amdgcn_s_wait_tensorcnt(0);
    }
    __syncthreads();
#pragma unroll
    for (int c2 = 0; c2 < 2; ++c2) {
      v16bf af[2], bfr[4];
#pragma unroll
      for (int i = 0; i < 2; ++i)
        af[i] = ld_frag(&As[(wm*32 + i*16 + lidx)*72 + c2*32 + hf*8]);
#pragma unroll
      for (int j = 0; j < 4; ++j)
        bfr[j] = ld_frag(&Bs[(wn*64 + j*16 + lidx)*72 + c2*32 + hf*8]);
#pragma unroll
      for (int i = 0; i < 2; ++i)
#pragma unroll
        for (int j = 0; j < 4; ++j)
          acc[i][j] = wmma_bf16(af[i], bfr[j], acc[i][j]);
    }
    __syncthreads();
  }

  // Epilogue. D-frag: element r -> row = r + 8*half, col = lane&15.
#pragma unroll
  for (int i = 0; i < 2; ++i) {
#pragma unroll
    for (int j = 0; j < 4; ++j) {
      const int col = n0 + wn*64 + j*16 + lidx;
      const float bv = bias[col];
      const int rowb = m0 + wm*32 + i*16 + hf*8;
#pragma unroll
      for (int r = 0; r < 8; ++r) {
        const int row = rowb + r;
        const float val = acc[i][j][r] + bv;
        if (MODE == 0) {
          const int which = col >> 11;            // 0:Q 1:K 2:V
          const int d = col & 2047;
          const int h = d >> 7, hd = d & 127;
          const int b = row >> 11, s = row & 2047;
          if (which == 0)
            Qb[((size_t)(b*NHEAD + h)*SEQ + s)*HD + hd] =
                (__bf16)(val * 0.08838834764831845f);       // 1/sqrt(128)
          else if (which == 1)
            Kb[((size_t)(b*NHEAD + h)*SEQ + s)*HD + hd] = (__bf16)val;
          else                                              // V stored transposed
            Vt[((size_t)(b*NHEAD + h)*HD + hd)*SEQ + s] = (__bf16)val;
        } else {
          outf[(size_t)row * N + col] = val;
        }
      }
    }
  }
}

// ---------------------------------------------------------------------------
// Causal flash attention, one block = 128 q-rows of one (b,h).
// 8 waves; wave w owns 16 q-rows. k-tiles of 32 keys, TDM-staged K and V^T.
// ---------------------------------------------------------------------------
__global__ __launch_bounds__(256) void attn_kernel(
    const __bf16* __restrict__ Qb, const __bf16* __restrict__ Kb,
    const __bf16* __restrict__ Vt, __bf16* __restrict__ Ob) {
  __shared__ __bf16 Ks[32 * 136];       // [key][hd], TDM-padded stride 136
  __shared__ __bf16 Vs[128 * 40];       // V^T: [hd][key], TDM-padded stride 40
  __shared__ __bf16 Ps[8 * 16 * 40];    // per-wave P patch [16 rows][32 keys]
  const int tid  = threadIdx.x;
  const int lane = tid & 31, wv = tid >> 5;
  const int lidx = lane & 15, hf = lane >> 4;
  const int qt = blockIdx.x, bh = blockIdx.y;
  const __bf16* Qp  = Qb + (size_t)bh * SEQ * HD;
  const __bf16* Kp  = Kb + (size_t)bh * SEQ * HD;
  const __bf16* Vtp = Vt + (size_t)bh * HD * SEQ;
  const int qbase = qt*128 + wv*16;

  v16bf qf[4];                           // Q 16x128 resident as 4 A-frags
#pragma unroll
  for (int c = 0; c < 4; ++c)
    qf[c] = ld_frag(Qp + (size_t)(qbase + lidx)*HD + c*32 + hf*8);

  v8f o[8] = {};
  float rm[8], rl[8];
#pragma unroll
  for (int r = 0; r < 8; ++r) { rm[r] = -3.0e38f; rl[r] = 0.f; }

  __bf16* Pw = &Ps[wv * 16 * 40];
  const int nkt = (qt + 1) * 4;
  for (int kt = 0; kt < nkt; ++kt) {
    const int kb = kt * 32;
    if (wv == 0) {
      tdm_load_2d((unsigned)(uintptr_t)&Ks[0], Kp + (size_t)kb*HD,
                  128, 32, (unsigned long long)HD, 5, 3);   // 256B rows -> +16B
      tdm_load_2d((unsigned)(uintptr_t)&Vs[0], Vtp + kb,
                  32, 128, (unsigned long long)SEQ, 3, 3);  // 64B rows  -> +16B
      __builtin_amdgcn_s_wait_tensorcnt(0);
    }
    __syncthreads();
    if (kb <= qbase + 15) {              // wave-uniform -> EXEC all-ones for WMMA
      v8f s0 = {}, s1 = {};
#pragma unroll
      for (int c = 0; c < 4; ++c) {      // S = Q K^T over hd (K-dim = 128)
        v16bf k0f = ld_frag(&Ks[lidx*136        + c*32 + hf*8]);
        v16bf k1f = ld_frag(&Ks[(16 + lidx)*136 + c*32 + hf*8]);
        s0 = wmma_bf16(qf[c], k0f, s0);
        s1 = wmma_bf16(qf[c], k1f, s1);
      }
      float p0[8], p1[8], al[8];
#pragma unroll
      for (int r = 0; r < 8; ++r) {      // causal mask + online softmax per row
        const int q = qbase + r + hf*8;
        float v0 = (kb + lidx      <= q) ? s0[r] : -3.0e38f;
        float v1 = (kb + 16 + lidx <= q) ? s1[r] : -3.0e38f;
        float tm = fmaxf(v0, v1);
        tm = fmaxf(tm, __shfl_xor(tm, 1, 32));
        tm = fmaxf(tm, __shfl_xor(tm, 2, 32));
        tm = fmaxf(tm, __shfl_xor(tm, 4, 32));
        tm = fmaxf(tm, __shfl_xor(tm, 8, 32));
        const float nm = fmaxf(rm[r], tm);
        const float a  = __expf(rm[r] - nm);
        rm[r] = nm; al[r] = a;
        p0[r] = __expf(v0 - nm);
        p1[r] = __expf(v1 - nm);
        float ts = p0[r] + p1[r];
        ts += __shfl_xor(ts, 1, 32);
        ts += __shfl_xor(ts, 2, 32);
        ts += __shfl_xor(ts, 4, 32);
        ts += __shfl_xor(ts, 8, 32);
        rl[r] = rl[r]*a + ts;
      }
#pragma unroll
      for (int n = 0; n < 8; ++n)
#pragma unroll
        for (int r = 0; r < 8; ++r) o[n][r] *= al[r];
#pragma unroll
      for (int r = 0; r < 8; ++r) {      // D-layout -> A-layout via wave LDS patch
        Pw[(r + hf*8)*40 + lidx]      = (__bf16)p0[r];
        Pw[(r + hf*8)*40 + 16 + lidx] = (__bf16)p1[r];
      }
      v16bf pf = ld_frag(&Pw[lidx*40 + hf*8]);
#pragma unroll
      for (int n = 0; n < 8; ++n) {      // O += P V (K-dim = 32 keys)
        v16bf vfr = ld_frag(&Vs[(n*16 + lidx)*40 + hf*8]);
        o[n] = wmma_bf16(pf, vfr, o[n]);
      }
    }
    __syncthreads();
  }
  const int b = bh >> 4, h = bh & 15;    // merge heads -> [B,S,D] bf16
#pragma unroll
  for (int n = 0; n < 8; ++n)
#pragma unroll
    for (int r = 0; r < 8; ++r) {
      const int q = qbase + r + hf*8;
      const float val = o[n][r] / rl[r];
      Ob[((size_t)(b*SEQ + q))*D_MODEL + h*HD + n*16 + lidx] = (__bf16)val;
    }
}

// ---------------------------------------------------------------------------
// Launch
// ---------------------------------------------------------------------------
extern "C" void kernel_launch(void* const* d_in, const int* in_sizes, int n_in,
                              void* d_out, int out_size, void* d_ws, size_t ws_size,
                              hipStream_t stream) {
  (void)in_sizes; (void)n_in; (void)out_size; (void)ws_size;
  const float* hs    = (const float*)d_in[0];
  const float* gamma = (const float*)d_in[1];
  const float* beta  = (const float*)d_in[2];
  const float* Wqkv  = (const float*)d_in[3];
  const float* bqkv  = (const float*)d_in[4];
  const float* Wproj = (const float*)d_in[5];
  const float* bproj = (const float*)d_in[6];
  float* out = (float*)d_out;

  char* ws = (char*)d_ws;
  __bf16* Xb   = (__bf16*)(ws);                        // 8192x2048        (32 MB)
  __bf16* WqbT = (__bf16*)(ws + 33554432ull);          // [6144][2048]     (24 MB)
  __bf16* WpbT = (__bf16*)(ws + 58720256ull);          // [2048][2048]     ( 8 MB)
  __bf16* Qb   = (__bf16*)(ws + 67108864ull);          // [B,H,S,hd]       (32 MB)
  __bf16* Kb   = (__bf16*)(ws + 100663296ull);         // [B,H,S,hd]       (32 MB)
  __bf16* Vt   = (__bf16*)(ws + 134217728ull);         // [B,H,hd,S]       (32 MB)
  __bf16* Ob   = (__bf16*)(ws + 167772160ull);         // [B,S,D]          (32 MB)

  // 1) weights -> bf16, transposed so GEMM B tiles are straight TDM 2D tiles
  transpose_cvt_kernel<<<dim3(3*D_MODEL/32, D_MODEL/32), 256, 0, stream>>>(
      Wqkv, WqbT, D_MODEL, 3*D_MODEL);
  transpose_cvt_kernel<<<dim3(D_MODEL/32, D_MODEL/32), 256, 0, stream>>>(
      Wproj, WpbT, D_MODEL, D_MODEL);
  // 2) LayerNorm -> bf16
  ln_bf16_kernel<<<ROWS, 256, 0, stream>>>(hs, gamma, beta, Xb);
  // 3) QKV projection (WMMA + TDM), scatter Q/K and transposed V
  gemm_bf16_kernel<0><<<dim3(3*D_MODEL/128, ROWS/128), 256, 0, stream>>>(
      Xb, WqbT, bqkv, ROWS, 3*D_MODEL, D_MODEL, Qb, Kb, Vt, nullptr);
  // 4) causal flash attention (WMMA + TDM)
  attn_kernel<<<dim3(SEQ/128, BATCH*NHEAD), 256, 0, stream>>>(Qb, Kb, Vt, Ob);
  // 5) output projection (WMMA + TDM), fp32 out
  gemm_bf16_kernel<1><<<dim3(D_MODEL/128, ROWS/128), 256, 0, stream>>>(
      Ob, WpbT, bproj, ROWS, D_MODEL, D_MODEL, nullptr, nullptr, nullptr, out);
}
